// DCFormerBlock_35021163331637
// MI455X (gfx1250) — matrix operand order repaired and checked by reading, not compiled, not run on hardware
//
#include <hip/hip_runtime.h>

// ---------------------------------------------------------------------------
// DCFormer block for MI455X (gfx1250, wave32, WMMA).
// All large GEMMs run through v_wmma_f32_16x16x32_bf16 with fp32 accum,
// with double-buffered LDS tiles staged by GLOBAL_LOAD_ASYNC_TO_LDS_B128
// (ASYNCcnt) when the toolchain exposes the builtin.
// ---------------------------------------------------------------------------

typedef unsigned short u16;
typedef __bf16 bf16x16 __attribute__((ext_vector_type(16)));
typedef float f32x8 __attribute__((ext_vector_type(8)));

union Frag {
  bf16x16 v;
  uint4 q[2];
};

__device__ __forceinline__ u16 f32_to_bf16(float f) {
  unsigned int u = __float_as_uint(f);
  return (u16)((u + 0x7FFFu + ((u >> 16) & 1u)) >> 16);
}

// --- async global->LDS staging (gfx1250 async-tensor path), with fallback ---
#if defined(__HIP_DEVICE_COMPILE__) && \
    __has_builtin(__builtin_amdgcn_global_load_async_to_lds_b128) && \
    __has_builtin(__builtin_amdgcn_s_wait_asynccnt)
#define HAVE_ASYNC_LDS 1
#else
#define HAVE_ASYNC_LDS 0
#endif

// The builtin takes pointers to 16-byte int vectors: (global v4i*, lds v4i*, imm, imm).
typedef int v4i __attribute__((vector_size(16)));
typedef __attribute__((address_space(1))) v4i av4i_global;
typedef __attribute__((address_space(3))) v4i av4i_lds;

__device__ __forceinline__ void cp_async16(u16* ldst, const u16* gsrc) {
#if HAVE_ASYNC_LDS
  // as1 value == flat value for global; as3 offset == low 32 bits of flat LDS.
  __builtin_amdgcn_global_load_async_to_lds_b128(
      (av4i_global*)(uintptr_t)gsrc, (av4i_lds*)(uintptr_t)ldst, 0, 0);
#else
  *(uint4*)ldst = *(const uint4*)gsrc;
#endif
}

__device__ __forceinline__ void cp_async_wait() {
#if HAVE_ASYNC_LDS
  __builtin_amdgcn_s_wait_asynccnt(0);
#endif
}

// ---------------------------------------------------------------------------
// RMSNorm over rows of length D; writes f32 and bf16 copies.
// ---------------------------------------------------------------------------
__global__ __launch_bounds__(256) void rmsnorm_rows(
    const float* __restrict__ x, const float* __restrict__ w,
    float* __restrict__ y, u16* __restrict__ ybf, int D) {
  const int t = blockIdx.x;
  const int tid = threadIdx.x;
  const float* row = x + (long)t * D;
  float ss = 0.f;
  for (int c = tid; c < D; c += 256) {
    float v = row[c];
    ss += v * v;
  }
  __shared__ float red[256];
  red[tid] = ss;
  __syncthreads();
  for (int st = 128; st > 0; st >>= 1) {
    if (tid < st) red[tid] += red[tid + st];
    __syncthreads();
  }
  float inv = rsqrtf(red[0] / (float)D + 1e-6f);
  for (int c = tid; c < D; c += 256) {
    float v = row[c] * inv * w[c];
    if (y) y[(long)t * D + c] = v;
    if (ybf) ybf[(long)t * D + c] = f32_to_bf16(v);
  }
}

// ---------------------------------------------------------------------------
// Transpose (R x C, row stride ldin, f32) -> (C x R, bf16). 32x32 LDS tiles.
// ---------------------------------------------------------------------------
__global__ __launch_bounds__(256) void transpose_convert(
    const float* __restrict__ in, int ldin, int R, int C,
    u16* __restrict__ out) {
  __shared__ float tile[32][33];
  const int c0 = blockIdx.x * 32, r0 = blockIdx.y * 32;
  const int tx = threadIdx.x & 31, ty = threadIdx.x >> 5;
  for (int i = 0; i < 32; i += 8) {
    int r = r0 + ty + i, c = c0 + tx;
    tile[ty + i][tx] = (r < R && c < C) ? in[(long)r * ldin + c] : 0.f;
  }
  __syncthreads();
  for (int i = 0; i < 32; i += 8) {
    int c = c0 + ty + i, r = r0 + tx;
    if (c < C && r < R) out[(long)c * R + r] = f32_to_bf16(tile[tx][ty + i]);
  }
}

// ---------------------------------------------------------------------------
// bf16 WMMA GEMM: C(MxN,f32) = A(MxK,bf16,row-major) * Bt(NxK,bf16)^T
// Block tile 256x64, 8 waves (4 along M x 2 along N), wave tile 64x32 via
// 4x2 WMMA accumulators. LDS double-buffered; tiles staged asynchronously.
// K % 32 == 0, M % 256 == 0, N % 64 == 0. Batched via blockIdx.z.
// ---------------------------------------------------------------------------
#define LDSW 40  // padded LDS row stride in halves (keeps 16B alignment)

__global__ __launch_bounds__(256) void gemm_bf16(
    const u16* __restrict__ A, int lda, long aBatch,
    const u16* __restrict__ Bt, int ldb, long bBatch,
    float* __restrict__ C, u16* __restrict__ Cbf, int ldc, long cBatch,
    int K) {
  __shared__ alignas(16) u16 lA[2][256 * LDSW];
  __shared__ alignas(16) u16 lB[2][64 * LDSW];
  const int tid = threadIdx.x;
  const int lane = tid & 31;
  const int wave = tid >> 5;
  const int wm = wave & 3;   // 4 waves along M, 64 rows each
  const int wn = wave >> 2;  // 2 waves along N, 32 cols each
  const int bz = blockIdx.z;

  A += (long)bz * aBatch + (long)(blockIdx.y * 256) * lda;
  Bt += (long)bz * bBatch + (long)(blockIdx.x * 64) * ldb;
  const long cOff =
      (long)bz * cBatch + (long)(blockIdx.y * 256) * ldc + (long)blockIdx.x * 64;

  f32x8 acc[4][2] = {};
  const int aRow = tid >> 1;        // 0..127
  const int aCol = (tid & 1) * 16;  // 0 or 16 (halves)
  // CDNA5 16-bit A fragment swizzle: lanes 0-15 hold K {0..7,16..23} of row
  // M=lane; lanes 16-31 hold K {8..15,24..31} of row M=lane-16.
  const int klo = (lane >> 4) * 8;
  const int mrow = lane & 15;
  const int bks = (lane >> 4) * 8;  // B: same K swizzle as A (lanes 0-15 K {0..7,16..23}, lanes 16-31 K {8..15,24..31})

  const int nK = K >> 5;

  auto stage = [&](int buf, int k0) {
    // A tile: 256 rows x 32 halves; each thread copies rows aRow and aRow+128.
    const u16* sa = A + (long)aRow * lda + k0 + aCol;
    u16* da = &lA[buf][aRow * LDSW + aCol];
    cp_async16(da, sa);
    cp_async16(da + 8, sa + 8);
    const u16* sa2 = sa + 128L * lda;
    cp_async16(da + 128 * LDSW, sa2);
    cp_async16(da + 128 * LDSW + 8, sa2 + 8);
    // B tile: 64 rows x 32 halves; threads 0..127.
    if (tid < 128) {
      const u16* sb = Bt + (long)aRow * ldb + k0 + aCol;
      u16* db = &lB[buf][aRow * LDSW + aCol];
      cp_async16(db, sb);
      cp_async16(db + 8, sb + 8);
    }
  };

  stage(0, 0);
  cp_async_wait();
  __syncthreads();

  for (int kt = 0; kt < nK; ++kt) {
    const int cur = kt & 1;
    if (kt + 1 < nK) stage(1 - cur, (kt + 1) << 5);
    if (kt + 2 < nK)  // pull the tile after next toward L2 (global_prefetch_b8)
      __builtin_prefetch((const void*)(A + (long)aRow * lda + ((kt + 2) << 5) + aCol), 0, 1);

    Frag a[4], b[2];
#pragma unroll
    for (int mi = 0; mi < 4; ++mi) {
      const u16* p = &lA[cur][(wm * 64 + mi * 16 + mrow) * LDSW];
      a[mi].q[0] = *(const uint4*)(p + klo);
      a[mi].q[1] = *(const uint4*)(p + 16 + klo);
    }
#pragma unroll
    for (int ni = 0; ni < 2; ++ni) {
      const u16* p = &lB[cur][(wn * 32 + ni * 16 + mrow) * LDSW + bks];
      b[ni].q[0] = *(const uint4*)(p);
      b[ni].q[1] = *(const uint4*)(p + 16);
    }
#pragma unroll
    for (int mi = 0; mi < 4; ++mi)
#pragma unroll
      for (int ni = 0; ni < 2; ++ni)
        acc[mi][ni] = __builtin_amdgcn_wmma_f32_16x16x32_bf16(
            false, a[mi].v, false, b[ni].v, (short)0, acc[mi][ni], false, false);

    if (kt + 1 < nK) {
      cp_async_wait();   // next buffer's async copies landed in LDS
      __syncthreads();   // and everyone is done reading the current buffer
    }
  }

  // f32 C/D layout: lane 0-15 -> N=lane, regs r -> M=r; lanes 16-31 -> M=8+r.
  const int cRowBase = wm * 64 + ((lane >> 4) * 8);
  const int cCol = wn * 32 + (lane & 15);
#pragma unroll
  for (int mi = 0; mi < 4; ++mi) {
#pragma unroll
    for (int r = 0; r < 8; ++r) {
      int row = cRowBase + mi * 16 + r;
#pragma unroll
      for (int ni = 0; ni < 2; ++ni) {
        float v = acc[mi][ni][r];
        long idx = cOff + (long)row * ldc + ni * 16 + cCol;
        if (C) C[idx] = v;
        if (Cbf) Cbf[idx] = f32_to_bf16(v);
      }
    }
  }
}

// ---------------------------------------------------------------------------
// Per (t, head): RMSNorm(q,k) + RoPE + q-scaling; writes bf16 (N,T,Dh).
// Block = 128 threads = one (t, n). N=16, Dh=128 structural.
// ---------------------------------------------------------------------------
__global__ __launch_bounds__(128) void qk_norm_rope(
    const float* __restrict__ qkv, const float* __restrict__ qw,
    const float* __restrict__ kw, u16* __restrict__ qbf,
    u16* __restrict__ kbf, int T, int D) {
  const int n = blockIdx.x & 15;
  const int t = blockIdx.x >> 4;
  const int d = threadIdx.x;  // 0..127
  const float* base = qkv + (long)t * (3 * D) + n * 128;
  float qv = base[d];
  float kv = base[D + d];
  __shared__ float rq[128], rk[128], nq[128], nk[128];
  rq[d] = qv * qv;
  rk[d] = kv * kv;
  __syncthreads();
  for (int st = 64; st > 0; st >>= 1) {
    if (d < st) {
      rq[d] += rq[d + st];
      rk[d] += rk[d + st];
    }
    __syncthreads();
  }
  float qs = rsqrtf(rq[0] / 128.f + 1e-6f);
  float ks = rsqrtf(rk[0] / 128.f + 1e-6f);
  nq[d] = qv * qs * qw[d];
  nk[d] = kv * ks * kw[d];
  __syncthreads();
  if (d < 64) {
    float inv = powf(10000.0f, -(float)d / 64.0f);
    float ang = (float)t * inv;
    float c, s;
    sincosf(ang, &s, &c);
    float q1 = nq[d], q2 = nq[d + 64];
    float k1 = nk[d], k2 = nk[d + 64];
    const float scale = 0.08838834764831845f;  // 128^-0.5
    long ob = ((long)n * T + t) * 128;
    qbf[ob + 2 * d] = f32_to_bf16((q1 * c - q2 * s) * scale);
    qbf[ob + 2 * d + 1] = f32_to_bf16((q2 * c + q1 * s) * scale);
    kbf[ob + 2 * d] = f32_to_bf16(k1 * c - k2 * s);
    kbf[ob + 2 * d + 1] = f32_to_bf16(k2 * c + k1 * s);
  }
}

// ---------------------------------------------------------------------------
// Elementwise kernels.
// ---------------------------------------------------------------------------
__global__ __launch_bounds__(256) void gelu_inplace(float* g, long n) {
  for (long i = (long)blockIdx.x * 256 + threadIdx.x; i < n;
       i += (long)gridDim.x * 256) {
    float v = g[i];
    g[i] = 0.5f * v * (1.f + erff(v * 0.7071067811865475f));
  }
}

__global__ __launch_bounds__(256) void tanh_inplace(float* g, long n) {
  for (long i = (long)blockIdx.x * 256 + threadIdx.x; i < n;
       i += (long)gridDim.x * 256)
    g[i] = tanhf(g[i]);
}

__global__ __launch_bounds__(256) void silu_mul_bf(const float* __restrict__ a,
                                                   const float* __restrict__ b,
                                                   u16* __restrict__ o, long n) {
  for (long i = (long)blockIdx.x * 256 + threadIdx.x; i < n;
       i += (long)gridDim.x * 256) {
    float av = a[i];
    float sv = av / (1.f + __expf(-av));
    o[i] = f32_to_bf16(sv * b[i]);
  }
}

__global__ __launch_bounds__(256) void resid_add(const float* __restrict__ a,
                                                 const float* __restrict__ b,
                                                 float* __restrict__ o, long n) {
  for (long i = (long)blockIdx.x * 256 + threadIdx.x; i < n;
       i += (long)gridDim.x * 256)
    o[i] = a[i] + b[i];
}

// ---------------------------------------------------------------------------
// Dynamic-weight projection: w[t,c,i,n] = sum_k dwh[t,c*64+k]*qkw[c,k,i,n],
// rmsnorm_noscale over n for i<2, scatter into pre/post q/k arrays (T x 2 x 16).
// Block = 256 threads = one t (c:4, i:4, n:16).
// ---------------------------------------------------------------------------
__global__ __launch_bounds__(256) void dynw(
    const float* __restrict__ dwh, const float* __restrict__ qkw,
    float* __restrict__ preQw1, float* __restrict__ preQw2,
    float* __restrict__ preKw1, float* __restrict__ preKw2,
    float* __restrict__ postQw1, float* __restrict__ postQw2,
    float* __restrict__ postKw1, float* __restrict__ postKw2) {
  const int t = blockIdx.x, tid = threadIdx.x;
  const int n = tid & 15, i = (tid >> 4) & 3, c = tid >> 6;
  const float* h = dwh + (long)t * 256 + c * 64;
  float acc = 0.f;
  for (int k = 0; k < 64; ++k)
    acc += h[k] * qkw[(((c * 64 + k) * 4 + i) * 16) + n];
  __shared__ float ssum[256];
  ssum[tid] = acc * acc;
  __syncthreads();
  float out = acc;
  if (i < 2) {
    float s = 0.f;
    int base = tid & ~15;
    for (int j = 0; j < 16; ++j) s += ssum[base + j];
    out = acc * rsqrtf(s / 16.f + 1e-6f);
  }
  float* dst;
  switch (c) {
    case 0: dst = (i < 2) ? preQw1 : preQw2; break;
    case 1: dst = (i < 2) ? preKw1 : preKw2; break;
    case 2: dst = (i < 2) ? postQw1 : postQw2; break;
    default: dst = (i < 2) ? postKw1 : postKw2; break;
  }
  dst[(long)t * 32 + (i & 1) * 16 + n] = out;
}

// ---------------------------------------------------------------------------
// Fused: pre cross-head proj -> causal mask -> softmax (two-pass online)
// -> post cross-head proj -> bf16 probs. One block per query row t.
// ddv layout (T,64): [0:16)=pre_qdd [16:32)=pre_kdd [32:48)=post_qdd [48:64)=post_kdd
// ---------------------------------------------------------------------------
__global__ __launch_bounds__(256) void softmax_xhead(
    const float* __restrict__ logits, const float* __restrict__ preQw1,
    const float* __restrict__ preQw2, const float* __restrict__ preKw1,
    const float* __restrict__ preKw2, const float* __restrict__ postQw1,
    const float* __restrict__ postQw2, const float* __restrict__ postKw1,
    const float* __restrict__ postKw2, const float* __restrict__ ddv,
    u16* __restrict__ probs, int T) {
  const int t = blockIdx.x;
  const int tid = threadIdx.x;
  const long TT = (long)T * T;
  __shared__ float sQ1[32], sQ2[32], sPQ1[32], sPQ2[32];
  __shared__ float sQdd[16], sPQdd[16];
  __shared__ float red[256];
  __shared__ float sM[16], sL[16];
  if (tid < 32) {
    sQ1[tid] = preQw1[t * 32 + tid];
    sQ2[tid] = preQw2[t * 32 + tid];
    sPQ1[tid] = postQw1[t * 32 + tid];
    sPQ2[tid] = postQw2[t * 32 + tid];
  }
  if (tid < 16) {
    sQdd[tid] = ddv[t * 64 + tid];
    sPQdd[tid] = ddv[t * 64 + 32 + tid];
  }
  __syncthreads();

  float m[16], l[16];
#pragma unroll
  for (int n = 0; n < 16; ++n) {
    m[n] = -3.0e38f;
    l[n] = 0.f;
  }

  // pass 1: pre-proj + mask + online max/sum per head
  for (int s = tid; s < T; s += 256) {
    if (s > t) continue;
    float x[16];
#pragma unroll
    for (int n = 0; n < 16; ++n) x[n] = logits[(long)n * TT + (long)t * T + s];
    float qh0 = 0, qh1 = 0, kh0 = 0, kh1 = 0;
#pragma unroll
    for (int n = 0; n < 16; ++n) {
      qh0 += x[n] * sQ1[n];
      qh1 += x[n] * sQ1[16 + n];
      kh0 += x[n] * preKw1[s * 32 + n];
      kh1 += x[n] * preKw1[s * 32 + 16 + n];
    }
#pragma unroll
    for (int n = 0; n < 16; ++n) {
      float y = x[n] + qh0 * sQ2[n] + qh1 * sQ2[16 + n] +
                kh0 * preKw2[s * 32 + n] + kh1 * preKw2[s * 32 + 16 + n] +
                x[n] * sQdd[n] + x[n] * ddv[s * 64 + 16 + n];
      float mn = fmaxf(m[n], y);
      l[n] = l[n] * __expf(m[n] - mn) + __expf(y - mn);
      m[n] = mn;
    }
  }

  // block reduction of (max, sum) per head
  for (int n = 0; n < 16; ++n) {
    red[tid] = m[n];
    __syncthreads();
    for (int st = 128; st > 0; st >>= 1) {
      if (tid < st) red[tid] = fmaxf(red[tid], red[tid + st]);
      __syncthreads();
    }
    if (tid == 0) sM[n] = red[0];
    __syncthreads();
    red[tid] = l[n] * __expf(m[n] - sM[n]);
    __syncthreads();
    for (int st = 128; st > 0; st >>= 1) {
      if (tid < st) red[tid] += red[tid + st];
      __syncthreads();
    }
    if (tid == 0) sL[n] = red[0];
    __syncthreads();
  }

  // pass 2: recompute pre-proj, normalize, post-proj, write bf16
  for (int s = tid; s < T; s += 256) {
    float p[16];
    if (s > t) {
#pragma unroll
      for (int n = 0; n < 16; ++n) p[n] = 0.f;
    } else {
      float x[16];
#pragma unroll
      for (int n = 0; n < 16; ++n) x[n] = logits[(long)n * TT + (long)t * T + s];
      float qh0 = 0, qh1 = 0, kh0 = 0, kh1 = 0;
#pragma unroll
      for (int n = 0; n < 16; ++n) {
        qh0 += x[n] * sQ1[n];
        qh1 += x[n] * sQ1[16 + n];
        kh0 += x[n] * preKw1[s * 32 + n];
        kh1 += x[n] * preKw1[s * 32 + 16 + n];
      }
#pragma unroll
      for (int n = 0; n < 16; ++n) {
        float y = x[n] + qh0 * sQ2[n] + qh1 * sQ2[16 + n] +
                  kh0 * preKw2[s * 32 + n] + kh1 * preKw2[s * 32 + 16 + n] +
                  x[n] * sQdd[n] + x[n] * ddv[s * 64 + 16 + n];
        p[n] = __expf(y - sM[n]) / sL[n];
      }
    }
    // post cross-head projection on probabilities
    float qh0 = 0, qh1 = 0, kh0 = 0, kh1 = 0;
#pragma unroll
    for (int n = 0; n < 16; ++n) {
      qh0 += p[n] * sPQ1[n];
      qh1 += p[n] * sPQ1[16 + n];
      kh0 += p[n] * postKw1[s * 32 + n];
      kh1 += p[n] * postKw1[s * 32 + 16 + n];
    }
#pragma unroll
    for (int n = 0; n < 16; ++n) {
      float o = p[n] + qh0 * sPQ2[n] + qh1 * sPQ2[16 + n] +
                kh0 * postKw2[s * 32 + n] + kh1 * postKw2[s * 32 + 16 + n] +
                p[n] * sPQdd[n] + p[n] * ddv[s * 64 + 48 + n];
      probs[(long)n * TT + (long)t * T + s] = f32_to_bf16(o);
    }
  }
}

// ---------------------------------------------------------------------------
// Host-side orchestration.
// ---------------------------------------------------------------------------
extern "C" void kernel_launch(void* const* d_in, const int* in_sizes, int n_in,
                              void* d_out, int out_size, void* d_ws,
                              size_t ws_size, hipStream_t stream) {
  (void)n_in;
  (void)out_size;
  (void)ws_size;
  const float* x = (const float*)d_in[0];
  const float* attn_norm_w = (const float*)d_in[1];
  const float* wqkv = (const float*)d_in[2];
  const float* q_norm_w = (const float*)d_in[3];
  const float* k_norm_w = (const float*)d_in[4];
  const float* dw1 = (const float*)d_in[5];
  const float* qkw = (const float*)d_in[6];
  const float* dd = (const float*)d_in[7];
  const float* wo = (const float*)d_in[8];
  const float* ffn_norm_w = (const float*)d_in[9];
  const float* w1f = (const float*)d_in[10];
  const float* w3f = (const float*)d_in[11];
  const float* w2f = (const float*)d_in[12];
  float* out = (float*)d_out;

  const int D = in_sizes[1];       // 2048
  const int T = in_sizes[0] / D;   // 2048
  const int Dh = in_sizes[3];      // 128
  const int N = D / Dh;            // 16
  const int F = in_sizes[10] / D;  // 5632

  size_t cur = 0;
  auto wsAlloc = [&](size_t bytes) -> char* {
    char* p = (char*)d_ws + cur;
    cur = (cur + bytes + 255) & ~(size_t)255;
    return p;
  };

  float* hn = (float*)wsAlloc((size_t)T * D * 4);
  u16* hn_bf = (u16*)wsAlloc((size_t)T * D * 2);
  u16* wqkvT = (u16*)wsAlloc((size_t)3 * D * D * 2);
  u16* dw1T = (u16*)wsAlloc((size_t)256 * D * 2);
  u16* ddT = (u16*)wsAlloc((size_t)64 * D * 2);
  u16* woT = (u16*)wsAlloc((size_t)D * D * 2);
  u16* w1fT = (u16*)wsAlloc((size_t)F * D * 2);
  u16* w3fT = (u16*)wsAlloc((size_t)F * D * 2);
  u16* w2fT = (u16*)wsAlloc((size_t)D * F * 2);
  float* qkv = (float*)wsAlloc((size_t)T * 3 * D * 4);
  u16* q_bf = (u16*)wsAlloc((size_t)N * T * Dh * 2);
  u16* k_bf = (u16*)wsAlloc((size_t)N * T * Dh * 2);
  u16* vT_bf = (u16*)wsAlloc((size_t)N * Dh * T * 2);
  float* dwh = (float*)wsAlloc((size_t)T * 256 * 4);
  float* ddv = (float*)wsAlloc((size_t)T * 64 * 4);
  float* cw = (float*)wsAlloc((size_t)8 * T * 32 * 4);
  float* preQw1 = cw + 0L * T * 32;
  float* preQw2 = cw + 1L * T * 32;
  float* preKw1 = cw + 2L * T * 32;
  float* preKw2 = cw + 3L * T * 32;
  float* postQw1 = cw + 4L * T * 32;
  float* postQw2 = cw + 5L * T * 32;
  float* postKw1 = cw + 6L * T * 32;
  float* postKw2 = cw + 7L * T * 32;
  u16* o_bf = (u16*)wsAlloc((size_t)T * D * 2);
  size_t logitsOff = cur;
  float* logits = (float*)wsAlloc((size_t)N * T * T * 4);
  size_t probsOff = cur;
  u16* probs_bf = (u16*)wsAlloc((size_t)N * T * T * 2);

  // FFN intermediates alias the (consumed-by-then) logits region.
  float* g1 = (float*)((char*)d_ws + logitsOff);
  float* g3 = g1 + (size_t)T * F;
  u16* gate_bf = (u16*)(g3 + (size_t)T * F);
  float* ffn = (float*)((char*)gate_bf + (size_t)T * F * 2);
  // h / fn alias the (consumed-by-then) probs region.
  float* proj = (float*)((char*)d_ws + probsOff);
  float* h = proj + (size_t)T * D;
  float* fn = h + (size_t)T * D;
  u16* fn_bf = (u16*)(fn + (size_t)T * D);

  // 1) attention RMSNorm
  rmsnorm_rows<<<T, 256, 0, stream>>>(x, attn_norm_w, hn, hn_bf, D);

  // 2) weight transpose+convert to N-major bf16
  transpose_convert<<<dim3((3 * D) / 32, D / 32), 256, 0, stream>>>(wqkv, 3 * D, D, 3 * D, wqkvT);
  transpose_convert<<<dim3(256 / 32, D / 32), 256, 0, stream>>>(dw1, 256, D, 256, dw1T);
  transpose_convert<<<dim3(64 / 32, D / 32), 256, 0, stream>>>(dd, 64, D, 64, ddT);
  transpose_convert<<<dim3(D / 32, D / 32), 256, 0, stream>>>(wo, D, D, D, woT);
  transpose_convert<<<dim3(F / 32, D / 32), 256, 0, stream>>>(w1f, F, D, F, w1fT);
  transpose_convert<<<dim3(F / 32, D / 32), 256, 0, stream>>>(w3f, F, D, F, w3fT);
  transpose_convert<<<dim3(D / 32, F / 32), 256, 0, stream>>>(w2f, D, F, D, w2fT);

  // 3) QKV projection
  gemm_bf16<<<dim3((3 * D) / 64, T / 256, 1), 256, 0, stream>>>(
      hn_bf, D, 0, wqkvT, D, 0, qkv, nullptr, 3 * D, 0, D);

  // 4) q/k norm + rope + scale; v transpose to (N,Dh,T) bf16
  qk_norm_rope<<<T * N, 128, 0, stream>>>(qkv, q_norm_w, k_norm_w, q_bf, k_bf, T, D);
  transpose_convert<<<dim3(D / 32, T / 32), 256, 0, stream>>>(qkv + 2 * D, 3 * D, T, D, vT_bf);

  // 5) dynamic weight path
  gemm_bf16<<<dim3(256 / 64, T / 256, 1), 256, 0, stream>>>(
      hn_bf, D, 0, dw1T, D, 0, dwh, nullptr, 256, 0, D);
  gelu_inplace<<<1024, 256, 0, stream>>>(dwh, (long)T * 256);
  gemm_bf16<<<dim3(1, T / 256, 1), 256, 0, stream>>>(
      hn_bf, D, 0, ddT, D, 0, ddv, nullptr, 64, 0, D);
  tanh_inplace<<<512, 256, 0, stream>>>(ddv, (long)T * 64);
  dynw<<<T, 256, 0, stream>>>(dwh, qkw, preQw1, preQw2, preKw1, preKw2,
                              postQw1, postQw2, postKw1, postKw2);

  // 6) per-head logits = q k^T (batched)
  gemm_bf16<<<dim3(T / 64, T / 256, N), 256, 0, stream>>>(
      q_bf, Dh, (long)T * Dh, k_bf, Dh, (long)T * Dh, logits, nullptr, T,
      (long)T * T, Dh);

  // 7) fused pre-proj + mask + softmax + post-proj -> bf16 probs
  softmax_xhead<<<T, 256, 0, stream>>>(logits, preQw1, preQw2, preKw1, preKw2,
                                       postQw1, postQw2, postKw1, postKw2, ddv,
                                       probs_bf, T);

  // 8) per-head o = probs @ v, written interleaved as (T, D) bf16
  gemm_bf16<<<dim3(Dh / 64, T / 256, N), 256, 0, stream>>>(
      probs_bf, T, (long)T * T, vT_bf, T, (long)Dh * T, nullptr, o_bf, D,
      (long)Dh, T);

  // 9) output projection + residual
  gemm_bf16<<<dim3(D / 64, T / 256, 1), 256, 0, stream>>>(
      o_bf, D, 0, woT, D, 0, proj, nullptr, D, 0, D);
  resid_add<<<2048, 256, 0, stream>>>(x, proj, h, (long)T * D);

  // 10) FFN
  rmsnorm_rows<<<T, 256, 0, stream>>>(h, ffn_norm_w, fn, fn_bf, D);
  gemm_bf16<<<dim3(F / 64, T / 256, 1), 256, 0, stream>>>(
      fn_bf, D, 0, w1fT, D, 0, g1, nullptr, F, 0, D);
  gemm_bf16<<<dim3(F / 64, T / 256, 1), 256, 0, stream>>>(
      fn_bf, D, 0, w3fT, D, 0, g3, nullptr, F, 0, D);
  silu_mul_bf<<<4096, 256, 0, stream>>>(g1, g3, gate_bf, (long)T * F);
  gemm_bf16<<<dim3(D / 64, T / 256, 1), 256, 0, stream>>>(
      gate_bf, F, 0, w2fT, F, 0, ffn, nullptr, D, 0, F);
  resid_add<<<2048, 256, 0, stream>>>(h, ffn, out, (long)T * D);
}
